// MultiHeadAttention_55070070670103
// MI455X (gfx1250) — matrix operand order repaired and checked by reading, not compiled, not run on hardware
//
#include <hip/hip_runtime.h>
#include <hip/hip_bf16.h>

// ---------------------------------------------------------------------------
// MHA forward for MI455X (gfx1250, wave32, WMMA).
// cast -> 3x bf16 GEMM (TDM-staged, double-buffered) -> RoPE/head-reshape ->
// flash attention (async-to-LDS staged, double-buffered, online softmax) ->
// bf16 GEMM -> fp32 out.
// ---------------------------------------------------------------------------

typedef __attribute__((ext_vector_type(16))) __bf16 v16bf;
typedef __attribute__((ext_vector_type(8)))  __bf16 v8bf;
typedef __attribute__((ext_vector_type(4)))  __bf16 v4bf;
typedef __attribute__((ext_vector_type(8)))  float  v8f;
typedef uint32_t u32x4 __attribute__((ext_vector_type(4)));
typedef uint32_t u32x8 __attribute__((ext_vector_type(8)));

union BF16x16 { v16bf v; v8bf h[2]; };

__device__ __forceinline__ v8f vzero8() {
    v8f z;
#pragma unroll
    for (int i = 0; i < 8; ++i) z[i] = 0.f;
    return z;
}

static constexpr int Bc  = 4;
static constexpr int Tc  = 2048;
static constexpr int Dc  = 2048;
static constexpr int Hc  = 16;
static constexpr int HDc = 128;
static constexpr int BTc = Bc * Tc;   // 8192

// -------------------- CDNA5 data-movement helpers --------------------------

// Low 32 bits of a generic pointer to LDS == LDS byte offset (aperture rule).
__device__ __forceinline__ uint32_t lds_off(const void* p) {
    return (uint32_t)(uintptr_t)p;
}

// TDM 2-D tile load: tile_w elements (2B each) x tile_h rows, global row
// stride row_stride elements, LDS padding pad_amt(enc) dwords after every
// 2^(pad_int+1) dwords.  D# group0 in 4 SGPRs, group1 in 8 SGPRs; 2-D form
// (VADDR2/3 = NULL).  TENSORcnt tracks completion.
__device__ __forceinline__ void tdm_load_2d(uint32_t ldsaddr, const void* g,
                                            uint32_t tile_w, uint32_t tile_h,
                                            uint32_t row_stride,
                                            uint32_t pad_int, uint32_t pad_amt) {
    uint64_t ga = (uint64_t)(uintptr_t)g;
    u32x4 g0;
    g0[0] = 1u;                                    // count=1, user, load
    g0[1] = ldsaddr;                               // lds_addr
    g0[2] = (uint32_t)ga;                          // global_addr[31:0]
    g0[3] = ((uint32_t)(ga >> 32) & 0x01FFFFFFu)   // global_addr[56:32]
          | (2u << 30);                            // type=2 ("image")
    u32x8 g1;
    g1[0] = (1u << 16)                             // data_size = 2 bytes
          | (1u << 20)                             // pad_enable
          | (pad_int << 22) | (pad_amt << 25);
    g1[1] = (tile_w & 0xFFFFu) << 16;                      // tensor_dim0 lo
    g1[2] = (tile_w >> 16) | ((tile_h & 0xFFFFu) << 16);   // dim0 hi | dim1 lo
    g1[3] = (tile_h >> 16) | ((tile_w & 0xFFFFu) << 16);   // dim1 hi | tile_dim0
    g1[4] = (tile_h & 0xFFFFu);                            // tile_dim1, tile_dim2=0
    g1[5] = row_stride;                                    // tensor_dim0_stride lo
    g1[6] = 0u;
    g1[7] = 0u;
    asm volatile("tensor_load_to_lds %0, %1" :: "s"(g0), "s"(g1) : "memory");
}

// Async 16B global -> LDS copy (ASYNCcnt path).
__device__ __forceinline__ void async_ld_b128(uint32_t ldsaddr, const void* g) {
    asm volatile("global_load_async_to_lds_b128 %0, %1, off"
                 :: "v"(ldsaddr), "v"((uint64_t)(uintptr_t)g) : "memory");
}
__device__ __forceinline__ void wait_async_0() {
    asm volatile("s_wait_asynccnt 0" ::: "memory");
}

// ---------------------------------------------------------------------------
// fp32 -> bf16 cast, 4 elements / thread
// ---------------------------------------------------------------------------
__global__ void cast4_f32_bf16(const float4* __restrict__ src,
                               __bf16* __restrict__ dst, int n4) {
    int i = blockIdx.x * blockDim.x + threadIdx.x;
    if (i < n4) {
        float4 f = src[i];
        v4bf o;
        o[0] = (__bf16)f.x; o[1] = (__bf16)f.y;
        o[2] = (__bf16)f.z; o[3] = (__bf16)f.w;
        *(v4bf*)(dst + (size_t)i * 4) = o;
    }
}

// ---------------------------------------------------------------------------
// C[M,N] (fp32) = A[M,K] bf16 * Bw[N,K]^T bf16  (y = x @ W.T form).
// 128x128 tile, 8 waves (2Mx4N), 4x2 WMMA fragments per wave.
// A/B tiles staged by TDM (tensor_load_to_lds) with LDS row pad 64B->80B,
// double-buffered so DMA of tile k+1 overlaps WMMA of tile k.
// ---------------------------------------------------------------------------
__global__ __launch_bounds__(256) void gemm_bf16_nt(
    const __bf16* __restrict__ A, const __bf16* __restrict__ Bw,
    float* __restrict__ C, int M, int N, int K) {
    __shared__ __bf16 Al[2][128 * 40];
    __shared__ __bf16 Bl[2][128 * 40];

    const int tid  = threadIdx.x;
    const int lane = tid & 31, wave = tid >> 5;
    const int half = lane >> 4, lr = lane & 15;
    const int wm = (wave >> 2) * 64;
    const int wn = (wave & 3) * 32;
    const int m0 = blockIdx.y * 128, n0 = blockIdx.x * 128;

    const __bf16* Arow = A + (size_t)m0 * K;
    const __bf16* Brow = Bw + (size_t)n0 * K;

    v8f acc[4][2];
#pragma unroll
    for (int i = 0; i < 4; ++i)
#pragma unroll
        for (int j = 0; j < 2; ++j) acc[i][j] = vzero8();

    // prologue: DMA tile 0 (pad: 16 dwords (64B) -> enc 3; pad 4 dwords -> enc 3)
    if (wave == 0) tdm_load_2d(lds_off(&Al[0][0]), Arow, 32, 128, K, 3, 3);
    if (wave == 1) tdm_load_2d(lds_off(&Bl[0][0]), Brow, 32, 128, K, 3, 3);
    __builtin_amdgcn_s_wait_tensorcnt(0);
    __syncthreads();

    int cur = 0;
    for (int k0 = 0; k0 < K; k0 += 32) {
        const int nxt = cur ^ 1;
        if (k0 + 32 < K) {   // DMA next tile while computing this one
            if (wave == 0) tdm_load_2d(lds_off(&Al[nxt][0]), Arow + k0 + 32, 32, 128, K, 3, 3);
            if (wave == 1) tdm_load_2d(lds_off(&Bl[nxt][0]), Brow + k0 + 32, 32, 128, K, 3, 3);
        }
        const __bf16* Ac = &Al[cur][0];
        const __bf16* Bb = &Bl[cur][0];

        BF16x16 af[4], bfr[2];
#pragma unroll
        for (int mf = 0; mf < 4; ++mf) {
            int m = wm + mf * 16 + lr;
            af[mf].h[0] = *(const v8bf*)(Ac + m * 40 + half * 8);
            af[mf].h[1] = *(const v8bf*)(Ac + m * 40 + 16 + half * 8);
        }
#pragma unroll
        for (int nf = 0; nf < 2; ++nf) {
            int n = wn + nf * 16 + lr;
            bfr[nf].h[0] = *(const v8bf*)(Bb + n * 40 + half * 8);
            bfr[nf].h[1] = *(const v8bf*)(Bb + n * 40 + 16 + half * 8);
        }
#pragma unroll
        for (int mf = 0; mf < 4; ++mf)
#pragma unroll
            for (int nf = 0; nf < 2; ++nf)
                acc[mf][nf] = __builtin_amdgcn_wmma_f32_16x16x32_bf16(
                    false, af[mf].v, false, bfr[nf].v,
                    (short)0, acc[mf][nf], false, false);

        __builtin_amdgcn_s_wait_tensorcnt(0);
        __syncthreads();
        cur = nxt;
    }

#pragma unroll
    for (int mf = 0; mf < 4; ++mf)
#pragma unroll
        for (int nf = 0; nf < 2; ++nf)
#pragma unroll
            for (int j = 0; j < 8; ++j) {
                int row = m0 + wm + mf * 16 + half * 8 + j;
                int col = n0 + wn + nf * 16 + lr;
                C[(size_t)row * N + col] = acc[mf][nf][j];
            }
}

// ---------------------------------------------------------------------------
// RoPE + head split: fp32 Q/K/V [B*T][D] -> bf16 Qh/Kh [bh][t][hd], Vt [bh][hd][t]
// ---------------------------------------------------------------------------
__global__ void rope_reshape(const float* __restrict__ Qf,
                             const float* __restrict__ Kf,
                             const float* __restrict__ Vf,
                             const float* __restrict__ cosT,
                             const float* __restrict__ sinT,
                             __bf16* __restrict__ Qh,
                             __bf16* __restrict__ Kh,
                             __bf16* __restrict__ Vt) {
    size_t idx = (size_t)blockIdx.x * blockDim.x + threadIdx.x; // [b][t][h][hd]
    int hd = (int)(idx & 127);
    int h  = (int)((idx >> 7) & 15);
    int t  = (int)((idx >> 11) & 2047);
    int b  = (int)(idx >> 22);

    float c = cosT[t * HDc + hd];
    float s = sinT[t * HDc + hd];
    size_t pidx = (idx & ~(size_t)127) | (size_t)(hd ^ 64);
    float sign = (hd < 64) ? -1.f : 1.f;

    float q = Qf[idx], qp = Qf[pidx];
    float k = Kf[idx], kp = Kf[pidx];
    float v = Vf[idx];

    size_t bh   = (size_t)b * Hc + h;
    size_t o_th = (bh * Tc + t) * HDc + hd;
    Qh[o_th] = (__bf16)(q * c + sign * qp * s);
    Kh[o_th] = (__bf16)(k * c + sign * kp * s);
    Vt[(bh * HDc + hd) * Tc + t] = (__bf16)v;
}

// ---------------------------------------------------------------------------
// Flash attention, causal.  256 threads / block = 8 waves; block handles one
// (b*H+h, 128-row q tile); each wave owns 16 q rows (full score row in-wave).
// K/V tiles staged with global_load_async_to_lds_b128, double-buffered:
// next-tile DMA overlaps S-WMMA + softmax + PV-WMMA of the current tile.
// Dyn LDS layout (bf16 elems): Q @0, K @(1+buf)*TSZ, V @(3+buf)*TSZ, P @5*TSZ,
// each tile 128x136 (row pad 256B->272B).  All buffer addresses computed at
// runtime (no LDS-pointer constant arrays -> no addrspacecast initializers).
// ---------------------------------------------------------------------------
__global__ __launch_bounds__(256) void flash_attn_bf16(
    const __bf16* __restrict__ Qh, const __bf16* __restrict__ Kh,
    const __bf16* __restrict__ Vt, __bf16* __restrict__ attnO) {
    extern __shared__ __bf16 smem[];
    constexpr int TSZ = 128 * 136;           // elements per tile buffer

    const int tid  = threadIdx.x;
    const int lane = tid & 31, wave = tid >> 5;
    const int half = lane >> 4, lr = lane & 15;
    const int qt = blockIdx.x;
    const int bh = blockIdx.y;               // b*H + h
    const int q0 = qt * 128;
    const int b  = bh >> 4, h = bh & 15;
    const float scale = 0.08838834764831845f; // 1/sqrt(128)

    __bf16* Ql = smem;                        // tile 0
    __bf16* Pl = smem + 5 * TSZ;              // tile 5

    // staging geometry: thread copies 16B segment `seg` of rows rbase+16j
    const int rbase = tid >> 4;              // 0..15
    const int seg   = tid & 15;
    const __bf16* qSrc = Qh + ((size_t)bh * Tc + q0 + rbase) * HDc + seg * 8;
    const __bf16* kSrc = Kh + ((size_t)bh * Tc + rbase) * HDc + seg * 8;
    const __bf16* vSrc = Vt + ((size_t)bh * HDc + rbase) * Tc + seg * 8;
    const uint32_t sbase = lds_off(smem);
    const uint32_t dOff  = (uint32_t)(rbase * 136 + seg * 8) * 2 + sbase;

    // stage Q + kv tile 0 asynchronously (K buf0 @ 1*TSZ, V buf0 @ 3*TSZ)
#pragma unroll
    for (int j = 0; j < 8; ++j)
        async_ld_b128(dOff + j * (16 * 136 * 2), qSrc + (size_t)j * 16 * HDc);
#pragma unroll
    for (int j = 0; j < 8; ++j) {
        async_ld_b128(dOff + 1 * TSZ * 2 + j * (16 * 136 * 2),
                      kSrc + (size_t)j * 16 * HDc);
        async_ld_b128(dOff + 3 * TSZ * 2 + j * (16 * 136 * 2),
                      vSrc + (size_t)j * 16 * Tc);
    }
    wait_async_0();
    __syncthreads();

    float mrow[8], lrow[8];
    v8f oacc[8];
#pragma unroll
    for (int j = 0; j < 8; ++j) { mrow[j] = -3.0e38f; lrow[j] = 0.f; }
#pragma unroll
    for (int nf = 0; nf < 8; ++nf) oacc[nf] = vzero8();

    int cur = 0;
    for (int kt = 0; kt <= qt; ++kt) {       // causal: upper tiles skipped
        const int kv0 = kt * 128;
        const int nxt = cur ^ 1;
        if (kt < qt) {                        // DMA next kv tile during compute
            const int kv1 = kv0 + 128;
            const uint32_t kD = dOff + (uint32_t)(1 + nxt) * (TSZ * 2);
            const uint32_t vD = dOff + (uint32_t)(3 + nxt) * (TSZ * 2);
#pragma unroll
            for (int j = 0; j < 8; ++j) {
                async_ld_b128(kD + j * (16 * 136 * 2),
                              kSrc + (size_t)(kv1 + j * 16) * HDc);
                async_ld_b128(vD + j * (16 * 136 * 2),
                              vSrc + (size_t)j * 16 * Tc + kv1);
            }
        }
        const __bf16* Kc = smem + (1 + cur) * TSZ;
        const __bf16* Vc = smem + (3 + cur) * TSZ;

        // ----- S = Q * K^T -----
        v8f sacc[8];
#pragma unroll
        for (int nf = 0; nf < 8; ++nf) sacc[nf] = vzero8();
#pragma unroll
        for (int ks = 0; ks < 4; ++ks) {
            BF16x16 a;
            int m = wave * 16 + lr;
            a.h[0] = *(const v8bf*)(Ql + m * 136 + ks * 32 + half * 8);
            a.h[1] = *(const v8bf*)(Ql + m * 136 + ks * 32 + 16 + half * 8);
#pragma unroll
            for (int nf = 0; nf < 8; ++nf) {
                BF16x16 bb;
                int n = nf * 16 + lr;
                bb.h[0] = *(const v8bf*)(Kc + n * 136 + ks * 32 + half * 8);
                bb.h[1] = *(const v8bf*)(Kc + n * 136 + ks * 32 + 16 + half * 8);
                sacc[nf] = __builtin_amdgcn_wmma_f32_16x16x32_bf16(
                    false, a.v, false, bb.v, (short)0, sacc[nf], false, false);
            }
        }

        // ----- scale + causal mask (diagonal tile only) -----
        const bool diag = (kt == qt);
#pragma unroll
        for (int nf = 0; nf < 8; ++nf)
#pragma unroll
            for (int j = 0; j < 8; ++j) {
                float s = sacc[nf][j] * scale;
                if (diag) {
                    int rg = q0 + wave * 16 + half * 8 + j;
                    int cg = kv0 + nf * 16 + lr;
                    if (cg > rg) s = -1e30f;
                }
                sacc[nf][j] = s;
            }

        // ----- online softmax -----
#pragma unroll
        for (int j = 0; j < 8; ++j) {
            float tmax = sacc[0][j];
#pragma unroll
            for (int nf = 1; nf < 8; ++nf) tmax = fmaxf(tmax, sacc[nf][j]);
#pragma unroll
            for (int off = 1; off < 16; off <<= 1)
                tmax = fmaxf(tmax, __shfl_xor(tmax, off, 32));

            float mn    = fmaxf(mrow[j], tmax);
            float alpha = __expf(mrow[j] - mn);
            mrow[j] = mn;

            float rs = 0.f;
#pragma unroll
            for (int nf = 0; nf < 8; ++nf) {
                float p = __expf(sacc[nf][j] - mn);
                sacc[nf][j] = p;
                rs += p;
            }
#pragma unroll
            for (int off = 1; off < 16; off <<= 1)
                rs += __shfl_xor(rs, off, 32);

            lrow[j] = lrow[j] * alpha + rs;
#pragma unroll
            for (int nf = 0; nf < 8; ++nf) oacc[nf][j] *= alpha;
        }

        // ----- P (C-layout) -> LDS row-major, for A-fragment reload -----
#pragma unroll
        for (int nf = 0; nf < 8; ++nf)
#pragma unroll
            for (int j = 0; j < 8; ++j)
                Pl[(wave * 16 + half * 8 + j) * 136 + nf * 16 + lr] =
                    (__bf16)sacc[nf][j];
        __syncthreads();

        // ----- O += P * V -----
#pragma unroll
        for (int ks = 0; ks < 4; ++ks) {
            BF16x16 a;
            int m = wave * 16 + lr;
            a.h[0] = *(const v8bf*)(Pl + m * 136 + ks * 32 + half * 8);
            a.h[1] = *(const v8bf*)(Pl + m * 136 + ks * 32 + 16 + half * 8);
#pragma unroll
            for (int nf = 0; nf < 8; ++nf) {
                BF16x16 bb;
                int n = nf * 16 + lr;
                bb.h[0] = *(const v8bf*)(Vc + n * 136 + ks * 32 + half * 8);
                bb.h[1] = *(const v8bf*)(Vc + n * 136 + ks * 32 + 16 + half * 8);
                oacc[nf] = __builtin_amdgcn_wmma_f32_16x16x32_bf16(
                    false, a.v, false, bb.v, (short)0, oacc[nf], false, false);
            }
        }

        wait_async_0();     // next-tile DMA done (own wave's ASYNCcnt)
        __syncthreads();    // ... and visible to every wave
        cur = nxt;
    }

    // ----- epilogue: normalize, merge heads -----
#pragma unroll
    for (int j = 0; j < 8; ++j) {
        float inv = 1.f / lrow[j];
        int rg = q0 + wave * 16 + half * 8 + j;
#pragma unroll
        for (int nf = 0; nf < 8; ++nf) {
            int col = h * HDc + nf * 16 + lr;
            attnO[((size_t)b * Tc + rg) * Dc + col] = (__bf16)(oacc[nf][j] * inv);
        }
    }
}

// ---------------------------------------------------------------------------
// Host launcher
// ---------------------------------------------------------------------------
extern "C" void kernel_launch(void* const* d_in, const int* in_sizes, int n_in,
                              void* d_out, int out_size, void* d_ws, size_t ws_size,
                              hipStream_t stream) {
    const float* x    = (const float*)d_in[0];
    const float* cosT = (const float*)d_in[1];
    const float* sinT = (const float*)d_in[2];
    // d_in[3] = additive causal mask: applied analytically, unused
    const float* Wq = (const float*)d_in[4];
    const float* Wk = (const float*)d_in[5];
    const float* Wv = (const float*)d_in[6];
    const float* Wo = (const float*)d_in[7];
    float* out = (float*)d_out;

    char* p = (char*)d_ws;
    auto carve = [&](size_t bytes) -> char* {
        char* r = p;
        p += (bytes + 255) & ~(size_t)255;
        return r;
    };
    const size_t nXD = (size_t)BTc * Dc;
    const size_t nW  = (size_t)Dc * Dc;
    __bf16* x_bf  = (__bf16*)carve(nXD * 2);
    __bf16* Wq_bf = (__bf16*)carve(nW * 2);
    __bf16* Wk_bf = (__bf16*)carve(nW * 2);
    __bf16* Wv_bf = (__bf16*)carve(nW * 2);
    __bf16* Wo_bf = (__bf16*)carve(nW * 2);
    float*  Qf    = (float*)carve(nXD * 4);
    float*  Kf    = (float*)carve(nXD * 4);
    float*  Vf    = (float*)carve(nXD * 4);
    __bf16* Qh    = (__bf16*)carve(nXD * 2);
    __bf16* Kh    = (__bf16*)carve(nXD * 2);
    __bf16* Vt    = (__bf16*)carve(nXD * 2);
    __bf16* attnO = (__bf16*)carve(nXD * 2);

    cast4_f32_bf16<<<(int)(nXD / 4 / 256), 256, 0, stream>>>((const float4*)x, x_bf, (int)(nXD / 4));
    cast4_f32_bf16<<<(int)(nW / 4 / 256), 256, 0, stream>>>((const float4*)Wq, Wq_bf, (int)(nW / 4));
    cast4_f32_bf16<<<(int)(nW / 4 / 256), 256, 0, stream>>>((const float4*)Wk, Wk_bf, (int)(nW / 4));
    cast4_f32_bf16<<<(int)(nW / 4 / 256), 256, 0, stream>>>((const float4*)Wv, Wv_bf, (int)(nW / 4));
    cast4_f32_bf16<<<(int)(nW / 4 / 256), 256, 0, stream>>>((const float4*)Wo, Wo_bf, (int)(nW / 4));

    dim3 gg(Dc / 128, BTc / 128);
    gemm_bf16_nt<<<gg, 256, 0, stream>>>(x_bf, Wq_bf, Qf, BTc, Dc, Dc);
    gemm_bf16_nt<<<gg, 256, 0, stream>>>(x_bf, Wk_bf, Kf, BTc, Dc, Dc);
    gemm_bf16_nt<<<gg, 256, 0, stream>>>(x_bf, Wv_bf, Vf, BTc, Dc, Dc);

    rope_reshape<<<(int)(nXD / 256), 256, 0, stream>>>(Qf, Kf, Vf, cosT, sinT, Qh, Kh, Vt);

    const int smem = 6 * 128 * 136 * 2;  // 208,896 B dynamic LDS
    (void)hipFuncSetAttribute((const void*)flash_attn_bf16,
                              hipFuncAttributeMaxDynamicSharedMemorySize, smem);
    dim3 fg(Tc / 128, Bc * Hc);
    flash_attn_bf16<<<fg, 256, smem, stream>>>(Qh, Kh, Vt, attnO);

    gemm_bf16_nt<<<gg, 256, 0, stream>>>(attnO, Wo_bf, out, BTc, Dc, Dc);
}
